// EGNNLayer_34591666602697
// MI455X (gfx1250) — compile-verified, matrix operand order checked by reading, not run on hardware
//
#include <hip/hip_runtime.h>

typedef __attribute__((ext_vector_type(16))) _Float16 v16h;
typedef __attribute__((ext_vector_type(8)))  float    v8f;

#define N_NODES 50000
#define N_EDGES 800000
#define FDIM    64
#define EFDIM   16
#define MSG_IN  145      // 2F + 1 + EF
#define MSG_K   160      // padded to 5*32
#define NODE_K  128

// ---- workspace layout (bytes) ----
// [0, 12.8MB)          m_sum  : N*64 f32
// [12.8MB, 13.0MB)     cnt    : N    f32
// [13,000,192 ...)     packed f16 weights (WMMA B-fragment layout)
#define WS_CNT_OFF     ((size_t)N_NODES * FDIM * 4)
#define WS_WEIGHTS_OFF ((size_t)13000192)

// half-element offsets inside the packed-weight region (each block = 512 halfs)
#define MW1B 0        // 5 kchunks * 4 ntiles
#define MW2B 10240    // 2 * 4
#define CW1B 14336    // 2 * 4
#define CW2B 18432    // 2 * 1
#define NW1B 19456    // 4 * 4
#define NW2B 27648    // 2 * 4

__device__ __forceinline__ v8f wmma16(v16h a, v16h b, v8f c) {
    return __builtin_amdgcn_wmma_f32_16x16x32_f16(
        /*neg_a=*/false, a, /*neg_b=*/false, b,
        /*c_mod=*/(short)0, c, /*reuse_a=*/false, /*reuse_b=*/false);
}

// Load one A-fragment row (16 halfs) from an LDS row, per ISA 16-bit A layout:
// lane<16 holds K = {0..7, 16..23} (+kcbase); lane>=16 holds K = {8..15, 24..31}.
__device__ __forceinline__ v16h load_a_row(const _Float16* rowptr, int kcbase, int khalf) {
    v16h a;
#pragma unroll
    for (int e = 0; e < 16; ++e) {
        int klocal = (e & 7) + 8 * khalf + 16 * (e >> 3);
        a[e] = rowptr[kcbase + klocal];
    }
    return a;
}

// ---------------- zero accumulators ----------------
__global__ __launch_bounds__(256) void egnn_zero(float* msum, float* cnt, float* pos_out) {
    int i = blockIdx.x * 256 + threadIdx.x;                 // grid covers N*64 exactly
    msum[i] = 0.0f;
    if (i < N_NODES)     cnt[i] = 0.0f;
    if (i < N_NODES * 3) pos_out[i] = 0.0f;
}

// ---------------- pack a weight matrix into WMMA B-fragment layout ----------------
__global__ __launch_bounds__(256) void egnn_prep_w(const float* __restrict__ W,
                                                   int Kact, int Nact, int ntiles,
                                                   int total, _Float16* __restrict__ out) {
    int idx = blockIdx.x * 256 + threadIdx.x;
    if (idx >= total) return;
    int block  = idx >> 9;            // 512 halfs per 32x16 block
    int within = idx & 511;
    int lane = within >> 4;
    int e    = within & 15;
    int kc = block / ntiles;
    int nt = block - kc * ntiles;
    int khalf  = lane >> 4;
    int coln   = lane & 15;
    int klocal = (e & 7) + 8 * khalf + 16 * (e >> 3);
    int k = kc * 32 + klocal;
    int n = nt * 16 + coln;
    float v = (k < Kact && n < Nact) ? W[(size_t)k * Nact + n] : 0.0f;
    out[idx] = (_Float16)v;
}

// ---------------- fused edge kernel ----------------
// 256 threads = 8 waves; each wave owns a 16-edge tile (128 edges / block).
__global__ __launch_bounds__(256) void egnn_edge(
    const float* __restrict__ x, const int* __restrict__ ei,
    const float* __restrict__ pos, const float* __restrict__ pos_init,
    const float* __restrict__ edge_attr,
    const float* __restrict__ mb1, const float* __restrict__ mb2,
    const float* __restrict__ cb1, const float* __restrict__ cb2,
    const _Float16* __restrict__ wsW,
    float* __restrict__ msum, float* __restrict__ cnt, float* __restrict__ pos_out)
{
    __shared__ _Float16 sfeat[128][MSG_K];   // 40 KB  f16 message features
    __shared__ _Float16 shid[8][16][FDIM];   // 16 KB  per-wave relayout bounce
    __shared__ float sdsq[128];
    __shared__ float sdpi[128][3];
    __shared__ float srinv[128];
    __shared__ int   ssrc[128];
    __shared__ int   sdst[128];

    const int tid   = threadIdx.x;
    const int ebase = blockIdx.x * 128;

    // phase 0: per-edge geometry + degree count
    if (tid < 128) {
        int e = ebase + tid;
        int s = ei[e];
        int d = ei[N_EDGES + e];
        ssrc[tid] = s; sdst[tid] = d;
        float dx = pos[3*s+0] - pos[3*d+0];
        float dy = pos[3*s+1] - pos[3*d+1];
        float dz = pos[3*s+2] - pos[3*d+2];
        sdsq[tid] = dx*dx + dy*dy + dz*dz;
        float ix = pos_init[3*s+0] - pos_init[3*d+0];
        float iy = pos_init[3*s+1] - pos_init[3*d+1];
        float iz = pos_init[3*s+2] - pos_init[3*d+2];
        sdpi[tid][0] = ix; sdpi[tid][1] = iy; sdpi[tid][2] = iz;
        srinv[tid] = 1.0f / sqrtf(ix*ix + iy*iy + iz*iz);   // no self loops
        atomicAdd(&cnt[d], 1.0f);
    }
    __syncthreads();

    // phase 1: build f16 feature tile [128, 160] (coalesced along k)
    for (int idx = tid; idx < 128 * MSG_K; idx += 256) {
        int row = idx / MSG_K, k = idx - row * MSG_K;
        float v;
        if      (k < FDIM)       v = x[(size_t)ssrc[row] * FDIM + k];
        else if (k < 2*FDIM)     v = x[(size_t)sdst[row] * FDIM + (k - FDIM)];
        else if (k == 2*FDIM)    v = sdsq[row];
        else if (k < MSG_IN)     v = edge_attr[(size_t)(ebase + row) * EFDIM + (k - 2*FDIM - 1)];
        else                     v = 0.0f;
        sfeat[row][k] = (_Float16)v;
    }
    __syncthreads();

    const int wave  = tid >> 5;
    const int lane  = tid & 31;
    const int col   = lane & 15;
    const int khalf = lane >> 4;
    const _Float16* arow = &sfeat[wave * 16 + col][0];

    // GEMM1: hidden = relu(feat @ mw1 + mb1)   (5 kchunks x 4 ntiles)
    v8f acc[4] = {v8f{}, v8f{}, v8f{}, v8f{}};
#pragma unroll
    for (int kc = 0; kc < 5; ++kc) {
        v16h a = load_a_row(arow, kc * 32, khalf);
#pragma unroll
        for (int nt = 0; nt < 4; ++nt) {
            v16h b = *(const v16h*)(wsW + MW1B + (kc * 4 + nt) * 512 + lane * 16);
            acc[nt] = wmma16(a, b, acc[nt]);
        }
    }
#pragma unroll
    for (int nt = 0; nt < 4; ++nt) {
        float bia = mb1[nt * 16 + col];
#pragma unroll
        for (int r = 0; r < 8; ++r) {
            float v = acc[nt][r] + bia;
            shid[wave][r + 8 * khalf][nt * 16 + col] = (_Float16)(v > 0.0f ? v : 0.0f);
        }
    }
    const _Float16* hrow = &shid[wave][col][0];

    // GEMM2: m_ij = hidden @ mw2 + mb2   (2 x 4)
    v8f mij[4] = {v8f{}, v8f{}, v8f{}, v8f{}};
#pragma unroll
    for (int kc = 0; kc < 2; ++kc) {
        v16h a = load_a_row(hrow, kc * 32, khalf);
#pragma unroll
        for (int nt = 0; nt < 4; ++nt) {
            v16h b = *(const v16h*)(wsW + MW2B + (kc * 4 + nt) * 512 + lane * 16);
            mij[nt] = wmma16(a, b, mij[nt]);
        }
    }
    // bias + scatter-add into m_sum, and stash m_ij as f16 for the coord MLP
#pragma unroll
    for (int nt = 0; nt < 4; ++nt) {
        float bia = mb2[nt * 16 + col];
#pragma unroll
        for (int r = 0; r < 8; ++r) {
            int row = r + 8 * khalf;
            float v = mij[nt][r] + bia;
            atomicAdd(&msum[(size_t)sdst[wave * 16 + row] * FDIM + nt * 16 + col], v);
            shid[wave][row][nt * 16 + col] = (_Float16)v;
        }
    }

    // GEMM3: t = relu(m_ij @ cw1 + cb1)   (2 x 4)
    v8f tac[4] = {v8f{}, v8f{}, v8f{}, v8f{}};
#pragma unroll
    for (int kc = 0; kc < 2; ++kc) {
        v16h a = load_a_row(hrow, kc * 32, khalf);
#pragma unroll
        for (int nt = 0; nt < 4; ++nt) {
            v16h b = *(const v16h*)(wsW + CW1B + (kc * 4 + nt) * 512 + lane * 16);
            tac[nt] = wmma16(a, b, tac[nt]);
        }
    }
#pragma unroll
    for (int nt = 0; nt < 4; ++nt) {
        float bia = cb1[nt * 16 + col];
#pragma unroll
        for (int r = 0; r < 8; ++r) {
            float v = tac[nt][r] + bia;
            shid[wave][r + 8 * khalf][nt * 16 + col] = (_Float16)(v > 0.0f ? v : 0.0f);
        }
    }

    // GEMM4: coord_w = t @ cw2 + cb2  (cw2 padded 64x16, only cols 0..2 live)
    v8f cac = v8f{};
#pragma unroll
    for (int kc = 0; kc < 2; ++kc) {
        v16h a = load_a_row(hrow, kc * 32, khalf);
        v16h b = *(const v16h*)(wsW + CW2B + kc * 512 + lane * 16);
        cac = wmma16(a, b, cac);
    }
    if (col < 3) {
        float bia = cb2[col];
#pragma unroll
        for (int r = 0; r < 8; ++r) {
            int el = wave * 16 + r + 8 * khalf;
            float w = cac[r] + bia;
            float v = sdpi[el][col] * srinv[el] * w;
            atomicAdd(&pos_out[(size_t)sdst[el] * 3 + col], v);
        }
    }
}

// ---------------- node kernel ----------------
__global__ __launch_bounds__(256) void egnn_node(
    const float* __restrict__ x, const float* __restrict__ msum,
    const float* __restrict__ cnt,
    const float* __restrict__ nb1, const float* __restrict__ nb2,
    const _Float16* __restrict__ wsW, float* __restrict__ hout)
{
    __shared__ _Float16 sfeat[128][NODE_K];  // 32 KB
    __shared__ _Float16 shid[8][16][FDIM];   // 16 KB
    __shared__ float sinv[128];

    const int tid   = threadIdx.x;
    const int nbase = blockIdx.x * 128;

    if (tid < 128) {
        int n = nbase + tid;
        float c = (n < N_NODES) ? cnt[n] : 1.0f;
        sinv[tid] = 1.0f / fmaxf(c, 1.0f);
    }
    __syncthreads();

    for (int idx = tid; idx < 128 * NODE_K; idx += 256) {
        int row = idx >> 7, k = idx & 127;
        int n = nbase + row;
        float v = 0.0f;
        if (n < N_NODES)
            v = (k < FDIM) ? x[(size_t)n * FDIM + k]
                           : msum[(size_t)n * FDIM + (k - FDIM)] * sinv[row];
        sfeat[row][k] = (_Float16)v;
    }
    __syncthreads();

    const int wave  = tid >> 5;
    const int lane  = tid & 31;
    const int col   = lane & 15;
    const int khalf = lane >> 4;
    const _Float16* arow = &sfeat[wave * 16 + col][0];

    // GEMM1: relu([x|m_i] @ nw1 + nb1)  (4 x 4)
    v8f acc[4] = {v8f{}, v8f{}, v8f{}, v8f{}};
#pragma unroll
    for (int kc = 0; kc < 4; ++kc) {
        v16h a = load_a_row(arow, kc * 32, khalf);
#pragma unroll
        for (int nt = 0; nt < 4; ++nt) {
            v16h b = *(const v16h*)(wsW + NW1B + (kc * 4 + nt) * 512 + lane * 16);
            acc[nt] = wmma16(a, b, acc[nt]);
        }
    }
#pragma unroll
    for (int nt = 0; nt < 4; ++nt) {
        float bia = nb1[nt * 16 + col];
#pragma unroll
        for (int r = 0; r < 8; ++r) {
            float v = acc[nt][r] + bia;
            shid[wave][r + 8 * khalf][nt * 16 + col] = (_Float16)(v > 0.0f ? v : 0.0f);
        }
    }
    const _Float16* hrow = &shid[wave][col][0];

    // GEMM2: h = hidden @ nw2 + nb2  (2 x 4)
    v8f hacc[4] = {v8f{}, v8f{}, v8f{}, v8f{}};
#pragma unroll
    for (int kc = 0; kc < 2; ++kc) {
        v16h a = load_a_row(hrow, kc * 32, khalf);
#pragma unroll
        for (int nt = 0; nt < 4; ++nt) {
            v16h b = *(const v16h*)(wsW + NW2B + (kc * 4 + nt) * 512 + lane * 16);
            hacc[nt] = wmma16(a, b, hacc[nt]);
        }
    }
#pragma unroll
    for (int nt = 0; nt < 4; ++nt) {
        float bia = nb2[nt * 16 + col];
#pragma unroll
        for (int r = 0; r < 8; ++r) {
            int n = nbase + wave * 16 + r + 8 * khalf;
            if (n < N_NODES)
                hout[(size_t)n * FDIM + nt * 16 + col] = hacc[nt][r] + bia;
        }
    }
}

extern "C" void kernel_launch(void* const* d_in, const int* in_sizes, int n_in,
                              void* d_out, int out_size, void* d_ws, size_t ws_size,
                              hipStream_t stream) {
    const float* x         = (const float*)d_in[0];
    const int*   ei        = (const int*)  d_in[1];
    const float* pos       = (const float*)d_in[2];
    const float* pos_init  = (const float*)d_in[3];
    const float* edge_attr = (const float*)d_in[4];
    const float* mw1 = (const float*)d_in[5];
    const float* mb1 = (const float*)d_in[6];
    const float* mw2 = (const float*)d_in[7];
    const float* mb2 = (const float*)d_in[8];
    const float* nw1 = (const float*)d_in[9];
    const float* nb1 = (const float*)d_in[10];
    const float* nw2 = (const float*)d_in[11];
    const float* nb2 = (const float*)d_in[12];
    const float* cw1 = (const float*)d_in[13];
    const float* cb1 = (const float*)d_in[14];
    const float* cw2 = (const float*)d_in[15];
    const float* cb2 = (const float*)d_in[16];

    float* h_out   = (float*)d_out;
    float* pos_out = h_out + (size_t)N_NODES * FDIM;

    float* msum = (float*)d_ws;
    float* cnt  = (float*)((char*)d_ws + WS_CNT_OFF);
    _Float16* wsW = (_Float16*)((char*)d_ws + WS_WEIGHTS_OFF);

    // zero accumulators (N*64 threads exactly)
    egnn_zero<<<(N_NODES * FDIM) / 256, 256, 0, stream>>>(msum, cnt, pos_out);

    // pack weights into WMMA B-fragment f16 layout
    egnn_prep_w<<<40, 256, 0, stream>>>(mw1, MSG_IN, 64, 4, 20 * 512, wsW + MW1B);
    egnn_prep_w<<<16, 256, 0, stream>>>(mw2, 64,     64, 4,  8 * 512, wsW + MW2B);
    egnn_prep_w<<<16, 256, 0, stream>>>(cw1, 64,     64, 4,  8 * 512, wsW + CW1B);
    egnn_prep_w<<< 4, 256, 0, stream>>>(cw2, 64,      3, 1,  2 * 512, wsW + CW2B);
    egnn_prep_w<<<32, 256, 0, stream>>>(nw1, 128,    64, 4, 16 * 512, wsW + NW1B);
    egnn_prep_w<<<16, 256, 0, stream>>>(nw2, 64,     64, 4,  8 * 512, wsW + NW2B);

    // fused edge pipeline: message MLP + scatter + coord MLP + coord scatter
    egnn_edge<<<N_EDGES / 128, 256, 0, stream>>>(
        x, ei, pos, pos_init, edge_attr, mb1, mb2, cb1, cb2,
        wsW, msum, cnt, pos_out);

    // node update MLP
    egnn_node<<<(N_NODES + 127) / 128, 256, 0, stream>>>(
        x, msum, cnt, nb1, nb2, wsW, h_out);
}
